// GraphConvolution_5746666242438
// MI455X (gfx1250) — matrix operand order repaired and checked by reading, not compiled, not run on hardware
//
#include <hip/hip_runtime.h>

#define N_NODES 40000
#define N_EDGES 640000
#define D_IN    512
#define D_OUT   128

typedef __attribute__((ext_vector_type(16))) __bf16       v16bf;
typedef __attribute__((ext_vector_type(8)))  float        v8f;
typedef __attribute__((ext_vector_type(4)))  unsigned int u32x4;
typedef __attribute__((ext_vector_type(8)))  int          i32x8;
typedef __attribute__((ext_vector_type(4)))  int          i32x4;

// Pre-split weight buffer: [kk][t][h(0=hi,1=lo)][lane][16 bf16]
//   element idx = (((kk*8 + t)*2 + h)*32 + lane)*16 + i
// Total 2*64K bf16 = 256 KB; each kk slice = 16 KB contiguous; TDM chunk = 4 kk = 64 KB.
#define WPK_ELEMS   (16 * 8 * 2 * 32 * 16)   // 131072 bf16
#define CHUNK_BYTES 65536                    // 4 kk-steps
#define NCHUNK      4

// ---------------------------------------------------------------------------
// Kernel 1: zero the aggregation target (d_out).
// ---------------------------------------------------------------------------
__global__ void __launch_bounds__(256) zero_kernel(float* __restrict__ out) {
    const int total4 = N_NODES * D_OUT / 4;
    int i = blockIdx.x * 256 + threadIdx.x;
    if (i < total4) {
        ((float4*)out)[i] = make_float4(0.f, 0.f, 0.f, 0.f);
    }
}

// ---------------------------------------------------------------------------
// Kernel 2: pre-split W (128x512 f32) into bf16 hi/lo in TDM/LDS-friendly
// fragment order (see WPK layout above).
//   column n = t*16 + (lane&15),  k = kk*32 + (lane>>4)*16 + i
// ---------------------------------------------------------------------------
__global__ void __launch_bounds__(256)
prep_weight(const float* __restrict__ W, __bf16* __restrict__ wpk) {
    int idx = blockIdx.x * 256 + threadIdx.x;   // 0 .. 65535
    int i    = idx & 15;
    int lane = (idx >> 4) & 31;
    int t    = (idx >> 9) & 7;
    int kk   = idx >> 12;

    int n = t * 16 + (lane & 15);
    int k = kk * 32 + (lane >> 4) * 16 + i;

    float f = W[n * D_IN + k];
    __bf16 h = (__bf16)f;
    int base = (((kk * 8 + t) * 2 + 0) * 32 + lane) * 16 + i;
    wpk[base]       = h;                         // hi slice
    wpk[base + 512] = (__bf16)(f - (float)h);    // lo slice (+32*16 elems)
}

// ---------------------------------------------------------------------------
// TDM: issue a 64 KB 1-D tile DMA  global -> LDS  (wave-level, EXEC ignored).
// D# built per CDNA5 ISA §8: data_size=8B, tile_dim0=tensor_dim0=8192, type=2.
// clang-23 toolchain: 6-arg builtin (u32x4, i32x8, i32x4, i32x4, i32x8, cpol).
// ---------------------------------------------------------------------------
__device__ __forceinline__ void tdm_load_chunk(const void* gsrc, unsigned lds_addr) {
    unsigned long long ga = (unsigned long long)(uintptr_t)gsrc;
    u32x4 g0;
    g0[0] = 1u;                                            // count=1, user mode
    g0[1] = lds_addr;                                      // LDS byte address
    g0[2] = (unsigned)(ga & 0xFFFFFFFFu);                  // global_addr[31:0]
    g0[3] = (unsigned)((ga >> 32) & 0x01FFFFFFu)           // global_addr[56:32]
            | (2u << 30);                                  // type=2 (image)
    i32x8 g1;
    g1[0] = (int)(3u << 16);          // workgroup_mask=0, data_size=3 (8 bytes)
    g1[1] = (int)(8192u << 16);       // tensor_dim0[15:0]=8192 in bits[63:48]
    g1[2] = (int)(1u << 16);          // tensor_dim0 hi=0, tensor_dim1[15:0]=1
    g1[3] = (int)(8192u << 16);       // tensor_dim1 hi=0, tile_dim0=8192
    g1[4] = 0;                        // tile_dim1=0, tile_dim2=0 (1-D tile)
    g1[5] = 8192;                     // tensor_dim0_stride[31:0]
    g1[6] = 0;
    g1[7] = 0;
    i32x4 z4 = {0, 0, 0, 0};
    i32x8 z8 = {0, 0, 0, 0, 0, 0, 0, 0};
    __builtin_amdgcn_tensor_load_to_lds(g0, g1, z4, z4, z8, 0);
}

// ---------------------------------------------------------------------------
// Kernel 3: seq = x @ W^T via bf16x2-split WMMA, B staged in LDS by the TDM.
//   grid = ceil(40000/128) blocks x 256 threads (8 wave32s), 128 KB dynamic LDS
//   wave w computes rows [blk*128 + w*16, +16) x all 128 cols (8 WMMA n-tiles)
// ---------------------------------------------------------------------------
__global__ void __launch_bounds__(256)
gemm_bf16x2_wmma(const float* __restrict__ X,
                 const __bf16* __restrict__ Wpk,   // pre-split W fragments (global)
                 float* __restrict__ seq) {
    extern __shared__ char smem[];                 // 2 x 64 KB double buffer

    const int lane = threadIdx.x & 31;
    const int wave = threadIdx.x >> 5;
    const int half = lane >> 4;
    const int r16  = lane & 15;

    const int m0 = blockIdx.x * 128 + wave * 16;

    int rowA = m0 + r16;
    if (rowA >= N_NODES) rowA = N_NODES - 1;
    const int khalfA = half * 16;

    const unsigned lds_base = (unsigned)(uintptr_t)smem;

    // Prologue: wave 0 kicks off chunk 0.
    if (wave == 0) {
        tdm_load_chunk((const char*)Wpk, lds_base);
    }

    v8f acc[8] = {};

    for (int c = 0; c < NCHUNK; ++c) {
        if (wave == 0) {
            if (c + 1 < NCHUNK) {
                // buf[(c+1)&1] was last read in chunk c-1; barrier at end of
                // that iteration guarantees it is free.
                tdm_load_chunk((const char*)Wpk + (size_t)(c + 1) * CHUNK_BYTES,
                               lds_base + ((unsigned)(c + 1) & 1u) * CHUNK_BYTES);
                __builtin_amdgcn_s_wait_tensorcnt(1);   // chunk c complete
            } else {
                __builtin_amdgcn_s_wait_tensorcnt(0);
            }
        }
        __syncthreads();                                // chunk c visible to all

        const char* bbase = smem + (c & 1) * CHUNK_BYTES;

        for (int kk2 = 0; kk2 < 4; ++kk2) {
            const int k = (c * 4 + kk2) * 32;

            // ---- A fragment: load f32, split into bf16 hi+lo ----
            float af[16];
            {
                const float* ap = X + (size_t)rowA * D_IN + (k + khalfA);
                *(float4*)(af + 0)  = *(const float4*)(ap + 0);
                *(float4*)(af + 4)  = *(const float4*)(ap + 4);
                *(float4*)(af + 8)  = *(const float4*)(ap + 8);
                *(float4*)(af + 12) = *(const float4*)(ap + 12);
            }
            v16bf a_hi, a_lo;
#pragma unroll
            for (int i = 0; i < 16; ++i) {
                __bf16 h = (__bf16)af[i];
                a_hi[i] = h;
                a_lo[i] = (__bf16)(af[i] - (float)h);
            }

#pragma unroll
            for (int t = 0; t < 8; ++t) {
                const int off = ((kk2 * 8 + t) * 2) * 1024 + lane * 32;
                v16bf b_hi = *(const v16bf*)(bbase + off);
                v16bf b_lo = *(const v16bf*)(bbase + off + 1024);

                // bf16x2 split: hi*hi + lo*hi + hi*lo (lo*lo negligible);
                // b_hi consumers adjacent to cover the b_lo LDS latency.
                acc[t] = __builtin_amdgcn_wmma_f32_16x16x32_bf16(
                             false, a_hi, false, b_hi, (short)0, acc[t], false, false);
                acc[t] = __builtin_amdgcn_wmma_f32_16x16x32_bf16(
                             false, a_lo, false, b_hi, (short)0, acc[t], false, false);
                acc[t] = __builtin_amdgcn_wmma_f32_16x16x32_bf16(
                             false, a_hi, false, b_lo, (short)0, acc[t], false, false);
            }
        }
        __syncthreads();   // all waves done reading buf[c&1] before it is refilled
    }

    // ---- Store: C/D layout — lane half selects M rows 0..7 / 8..15 ----
#pragma unroll
    for (int t = 0; t < 8; ++t) {
#pragma unroll
        for (int j = 0; j < 8; ++j) {
            int row = m0 + half * 8 + j;
            if (row < N_NODES)
                seq[(size_t)row * D_OUT + t * 16 + r16] = acc[t][j];
        }
    }
}

// ---------------------------------------------------------------------------
// Kernel 4: COO SpMM scatter — one wave per edge; lane handles 4 features.
// ---------------------------------------------------------------------------
__global__ void __launch_bounds__(256)
spmm_scatter(const float* __restrict__ seq,
             const float* __restrict__ ev,
             const int*   __restrict__ er,
             const int*   __restrict__ ec,
             float* __restrict__ agg) {
    const int e = blockIdx.x * 8 + (threadIdx.x >> 5);
    if (e >= N_EDGES) return;
    const int lane = threadIdx.x & 31;

    const int   c = ec[e];
    const int   r = er[e];
    const float v = ev[e];

    float4 m = ((const float4*)(seq + (size_t)c * D_OUT))[lane];
    float* dst = agg + (size_t)r * D_OUT + lane * 4;
    __hip_atomic_fetch_add(dst + 0, v * m.x, __ATOMIC_RELAXED, __HIP_MEMORY_SCOPE_AGENT);
    __hip_atomic_fetch_add(dst + 1, v * m.y, __ATOMIC_RELAXED, __HIP_MEMORY_SCOPE_AGENT);
    __hip_atomic_fetch_add(dst + 2, v * m.z, __ATOMIC_RELAXED, __HIP_MEMORY_SCOPE_AGENT);
    __hip_atomic_fetch_add(dst + 3, v * m.w, __ATOMIC_RELAXED, __HIP_MEMORY_SCOPE_AGENT);
}

// ---------------------------------------------------------------------------
// Kernel 5: out = PReLU(out + bias), in place on d_out.
// ---------------------------------------------------------------------------
__global__ void __launch_bounds__(256)
bias_prelu(float* __restrict__ out,
           const float* __restrict__ bias,
           const float* __restrict__ alpha) {
    const int total4 = N_NODES * D_OUT / 4;
    int i = blockIdx.x * 256 + threadIdx.x;
    if (i >= total4) return;

    const float a = alpha[0];
    float4 v = ((float4*)out)[i];
    float4 b = ((const float4*)bias)[i & (D_OUT / 4 - 1)];
    v.x += b.x; v.y += b.y; v.z += b.z; v.w += b.w;
    v.x = v.x > 0.f ? v.x : a * v.x;
    v.y = v.y > 0.f ? v.y : a * v.y;
    v.z = v.z > 0.f ? v.z : a * v.z;
    v.w = v.w > 0.f ? v.w : a * v.w;
    ((float4*)out)[i] = v;
}

// ---------------------------------------------------------------------------
extern "C" void kernel_launch(void* const* d_in, const int* in_sizes, int n_in,
                              void* d_out, int out_size, void* d_ws, size_t ws_size,
                              hipStream_t stream) {
    const float* x     = (const float*)d_in[0];   // [40000, 512]
    const float* w     = (const float*)d_in[1];   // [128, 512]
    const float* bias  = (const float*)d_in[2];   // [128]
    const float* alpha = (const float*)d_in[3];   // [1]
    const float* ev    = (const float*)d_in[4];   // [640000]
    const int*   er    = (const int*)d_in[5];     // [640000]
    const int*   ec    = (const int*)d_in[6];     // [640000]

    float* out = (float*)d_out;                   // [40000, 128]

    // Workspace: seq (20,480,000 B) then wpk (262,144 B, 8B-aligned).
    float*  seq = (float*)d_ws;
    __bf16* wpk = (__bf16*)((char*)d_ws + (size_t)N_NODES * D_OUT * sizeof(float));

    const int total4 = N_NODES * D_OUT / 4;       // 1,280,000

    zero_kernel<<<(total4 + 255) / 256, 256, 0, stream>>>(out);

    prep_weight<<<(WPK_ELEMS / 2) / 256, 256, 0, stream>>>(w, wpk);

    gemm_bf16x2_wmma<<<(N_NODES + 127) / 128, 256, 2 * CHUNK_BYTES, stream>>>(
        x, wpk, seq);

    spmm_scatter<<<(N_EDGES + 7) / 8, 256, 0, stream>>>(seq, ev, er, ec, out);

    bias_prelu<<<(total4 + 255) / 256, 256, 0, stream>>>(out, bias, alpha);
}
